// GraphEncoder_37769942401722
// MI455X (gfx1250) — compile-verified, matrix operand order checked by reading, not run on hardware
//
#include <hip/hip_runtime.h>
#include <hip/hip_bf16.h>

// ---- problem constants (match reference) ----
#define B_  8
#define N_  512
#define E_  8192
#define R_  8
#define L_  2
#define C_  1024
#define H_  8
#define Q_  10
#define HD_ 128           // C/H
#define BNC ((size_t)B_ * N_ * C_)

typedef __attribute__((ext_vector_type(16))) __bf16 v16bf;
typedef __attribute__((ext_vector_type(8)))  float  v8f;

// ---------------------------------------------------------------------------
// 1. Embedding gather: x[b,n,:] = emb_table[nodes[b,n], :]
// ---------------------------------------------------------------------------
__global__ void k_gather(const int* __restrict__ nodes,
                         const float* __restrict__ emb,
                         float* __restrict__ x) {
  int bn = blockIdx.x;
  int node = nodes[bn];
  const float4* s = (const float4*)(emb + (size_t)node * C_);
  float4* d = (float4*)(x + (size_t)bn * C_);
  d[threadIdx.x] = s[threadIdx.x];
}

// ---------------------------------------------------------------------------
// 2. Edge-count per (b, dst, r) and inverse normalization factor
// ---------------------------------------------------------------------------
__global__ void k_zero_i(int* p, int n) {
  int i = blockIdx.x * blockDim.x + threadIdx.x;
  if (i < n) p[i] = 0;
}

__global__ void k_count(const int* __restrict__ edges,
                        const int* __restrict__ types,
                        int* __restrict__ cnt) {
  int idx = blockIdx.x * blockDim.x + threadIdx.x;   // b*E + e
  if (idx >= B_ * E_) return;
  int b = idx / E_, e = idx - b * E_;
  int dst = edges[b * 2 * E_ + E_ + e];
  int r = types[idx];
  atomicAdd(&cnt[(b * N_ + dst) * R_ + r], 1);
}

__global__ void k_inv(const int* __restrict__ cnt, float* __restrict__ inv, int n) {
  int i = blockIdx.x * blockDim.x + threadIdx.x;
  if (i < n) {
    int c = cnt[i];
    inv[i] = 1.0f / (float)(c > 1 ? c : 1);
  }
}

// ---------------------------------------------------------------------------
// 3a. fp32 -> bf16 row convert (4 elements / thread)
// ---------------------------------------------------------------------------
__global__ void k_cvt4(const float* __restrict__ in, __bf16* __restrict__ out,
                       long n4) {
  long i = (long)blockIdx.x * blockDim.x + threadIdx.x;
  if (i >= n4) return;
  float4 a = ((const float4*)in)[i];
  union { __bf16 h[4]; unsigned long long u; } r;
  r.h[0] = (__bf16)a.x; r.h[1] = (__bf16)a.y;
  r.h[2] = (__bf16)a.z; r.h[3] = (__bf16)a.w;
  ((unsigned long long*)out)[i] = r.u;
}

// ---------------------------------------------------------------------------
// 3b. fp32 (K,N) -> bf16 (N,K) transpose-convert (C x C weight slice)
// ---------------------------------------------------------------------------
__global__ void k_cvtT(const float* __restrict__ in, __bf16* __restrict__ out) {
  int idx = blockIdx.x * blockDim.x + threadIdx.x;   // idx = n*C + k
  int k = idx & (C_ - 1);
  int n = idx >> 10;
  out[idx] = (__bf16)in[(size_t)k * C_ + n];
}

// ---------------------------------------------------------------------------
// 4. WMMA GEMM: Y[b](512x1024 f32) = A[b](512x1024 bf16) * Bt^T + bias
//    Bt is bf16 stored (N,K).  Block = 256 thr (8 waves), tile 128x128.
//    Double-buffered LDS B tile staged with global_load_async_to_lds_b128
//    (ASYNCcnt) overlapping the WMMA pipe; all 8 B fragments loaded as one
//    ds_load_b128 clause, then 8 back-to-back WMMAs per K-step.
//    grid = (M/128, batch, N/128)
// ---------------------------------------------------------------------------
__global__ void k_gemm_bf16(const __bf16* __restrict__ A,
                            const __bf16* __restrict__ Bt,
                            const float* __restrict__ biasv,
                            float* __restrict__ Y) {
  __shared__ __align__(16) __bf16 sB[2][128][32];    // 2 x 8 KB
  const int lane = threadIdx.x & 31;
  const int wave = threadIdx.x >> 5;
  const int bm = blockIdx.x * 128;
  const int bn = blockIdx.z * 128;
  const int half = lane >> 4;                        // K-group select
  const int mn = lane & 15;
  const int tm = bm + wave * 16;

  const __bf16* Ab = A + (size_t)blockIdx.y * N_ * C_;
  float*        Yb = Y + (size_t)blockIdx.y * N_ * C_;
  const __bf16* arow = Ab + (size_t)(tm + mn) * C_;

  // low 32 bits of a generic shared pointer == LDS byte offset
  unsigned sBbase = (unsigned)(size_t)&sB[0][0][0];

  // per-thread mapping of the 512 x 16B tile transfers (2 per thread)
  const int t0row = (threadIdx.x * 2) >> 2;          // local n of xfer 0
  const int t0seg = (threadIdx.x * 2) & 3;           // 16B segment of xfer 0

  auto stage = [&](int kstep, int bufi) {
#pragma unroll
    for (int i = 0; i < 2; ++i) {
      int t = threadIdx.x * 2 + i;
      int row = t >> 2;
      int seg = t & 3;
      unsigned lds = sBbase + bufi * 8192 + row * 64 + seg * 16;
      unsigned long long ga = (unsigned long long)(size_t)(
          Bt + (size_t)(bn + row) * C_ + kstep + seg * 8);
      asm volatile("global_load_async_to_lds_b128 %0, %1, off"
                   :: "v"(lds), "v"(ga) : "memory");
    }
  };
  (void)t0row; (void)t0seg;

  v8f acc[8];
#pragma unroll
  for (int nt = 0; nt < 8; ++nt) acc[nt] = (v8f){};

  // prologue: stage first B tile
  stage(0, 0);
  asm volatile("s_wait_asynccnt 0" ::: "memory");
  __syncthreads();

  for (int k0 = 0; k0 < C_; k0 += 32) {
    const int cur = (k0 >> 5) & 1;
    // prefetch next B tile into the other buffer (overlaps compute)
    if (k0 + 32 < C_) stage(k0 + 32, cur ^ 1);

    // A fragment: two contiguous 16B global loads (ISA 16-bit A layout)
    union { uint4 u[2]; v16bf v; } af;
    af.u[0] = *(const uint4*)(arow + k0 + half * 8);
    af.u[1] = *(const uint4*)(arow + k0 + half * 8 + 16);

    // load all 8 B fragments first (one ds clause, one wait)
    union { uint4 u[2]; v16bf v; } bf8[8];
#pragma unroll
    for (int nt = 0; nt < 8; ++nt) {
      const __bf16* brow = &sB[cur][nt * 16 + mn][half * 8];
      bf8[nt].u[0] = *(const uint4*)(brow);
      bf8[nt].u[1] = *(const uint4*)(brow + 16);
    }

    // 8 back-to-back WMMAs sharing the A fragment
#pragma unroll
    for (int nt = 0; nt < 8; ++nt) {
      acc[nt] = __builtin_amdgcn_wmma_f32_16x16x32_bf16(
          false, af.v, false, bf8[nt].v, (short)0, acc[nt], false, false);
    }

    // next tile arrived + all waves done reading current buffer
    asm volatile("s_wait_asynccnt 0" ::: "memory");
    __syncthreads();
  }

#pragma unroll
  for (int nt = 0; nt < 8; ++nt) {
    int n = bn + nt * 16 + mn;
    float bv = biasv ? biasv[n] : 0.0f;
#pragma unroll
    for (int j = 0; j < 8; ++j) {                    // lanes>=16 -> M = j+8
      Yb[(size_t)(tm + j + 8 * half) * C_ + n] = acc[nt][j] + bv;
    }
  }
}

// ---------------------------------------------------------------------------
// 5. Edge scatter for relation r (global fp32 atomics)
// ---------------------------------------------------------------------------
__global__ void k_scatter(const float* __restrict__ xt,
                          const int* __restrict__ edges,
                          const int* __restrict__ types,
                          const float* __restrict__ inv,
                          float* __restrict__ out,
                          int r) {
  int e = blockIdx.x, b = blockIdx.y;
  if (types[b * E_ + e] != r) return;
  int src = edges[b * 2 * E_ + e];
  int dst = edges[b * 2 * E_ + E_ + e];
  float s = inv[(b * N_ + dst) * R_ + r];
  const float4* xs = (const float4*)(xt + (size_t)(b * N_ + src) * C_);
  float* od = out + (size_t)(b * N_ + dst) * C_;
  float4 v = xs[threadIdx.x];
  int i = threadIdx.x * 4;
  atomicAdd(&od[i + 0], v.x * s);
  atomicAdd(&od[i + 1], v.y * s);
  atomicAdd(&od[i + 2], v.z * s);
  atomicAdd(&od[i + 3], v.w * s);
}

// ---------------------------------------------------------------------------
// 6. in-place ReLU
// ---------------------------------------------------------------------------
__global__ void k_relu(float* p, long n) {
  long i = (long)blockIdx.x * blockDim.x + threadIdx.x;
  if (i < n) p[i] = fmaxf(p[i], 0.0f);
}

// ---------------------------------------------------------------------------
// 7. q = gprompt @ Wq.T + bq   (10x1024, tiny)
// ---------------------------------------------------------------------------
__global__ void k_qproj(const float* __restrict__ gprompt,
                        const float* __restrict__ ipw,
                        const float* __restrict__ ipb,
                        float* __restrict__ q) {
  int idx = blockIdx.x * blockDim.x + threadIdx.x;    // j*C + d
  if (idx >= Q_ * C_) return;
  int d = idx & (C_ - 1);
  int j = idx >> 10;
  const float* g = gprompt + (size_t)j * C_;
  const float* w = ipw + (size_t)d * C_;
  float s = ipb[d];
  for (int c = 0; c < C_; ++c) s += g[c] * w[c];
  q[idx] = s;
}

// ---------------------------------------------------------------------------
// 8. scores[b,h,j,n] = (q_h[j] . k_h[b,n]) / sqrt(HD)
// ---------------------------------------------------------------------------
__global__ void k_scores(const float* __restrict__ q,
                         const float* __restrict__ kbuf,
                         float* __restrict__ scores) {
  int idx = blockIdx.x * blockDim.x + threadIdx.x;
  if (idx >= B_ * H_ * Q_ * N_) return;
  int n = idx & (N_ - 1);
  int t = idx >> 9;
  int j = t % Q_;  t /= Q_;
  int h = t & (H_ - 1);
  int b = t >> 3;
  const float* qp = q + (size_t)j * C_ + h * HD_;
  const float* kp = kbuf + (size_t)(b * N_ + n) * C_ + h * HD_;
  float s = 0.0f;
#pragma unroll 4
  for (int d = 0; d < HD_; ++d) s += qp[d] * kp[d];
  scores[idx] = s * 0.08838834764831845f;             // 1/sqrt(128)
}

// ---------------------------------------------------------------------------
// 9. softmax over N per (b,h,j) row; one wave32 per row
// ---------------------------------------------------------------------------
__global__ void k_softmax(float* __restrict__ scores) {
  int gtid = blockIdx.x * blockDim.x + threadIdx.x;
  int row = gtid >> 5;
  int lane = threadIdx.x & 31;
  if (row >= B_ * H_ * Q_) return;
  float* p = scores + (size_t)row * N_;
  float m = -3.402823e38f;
  for (int i = lane; i < N_; i += 32) m = fmaxf(m, p[i]);
#pragma unroll
  for (int o = 16; o > 0; o >>= 1) m = fmaxf(m, __shfl_xor(m, o, 32));
  float s = 0.0f;
  for (int i = lane; i < N_; i += 32) {
    float ev = __expf(p[i] - m);
    p[i] = ev;
    s += ev;
  }
#pragma unroll
  for (int o = 16; o > 0; o >>= 1) s += __shfl_xor(s, o, 32);
  float is = 1.0f / s;
  for (int i = lane; i < N_; i += 32) p[i] *= is;
}

// ---------------------------------------------------------------------------
// 10. o[b,j,h*HD+d] = sum_n attn[b,h,j,n] * v[b,n,h*HD+d]
// ---------------------------------------------------------------------------
__global__ void k_attnv(const float* __restrict__ attn,
                        const float* __restrict__ vbuf,
                        float* __restrict__ o) {
  int idx = blockIdx.x * blockDim.x + threadIdx.x;
  if (idx >= B_ * H_ * Q_ * HD_) return;
  int d = idx & (HD_ - 1);
  int t = idx >> 7;
  int j = t % Q_;  t /= Q_;
  int h = t & (H_ - 1);
  int b = t >> 3;
  const float* at = attn + (size_t)((b * H_ + h) * Q_ + j) * N_;
  const float* vp = vbuf + (size_t)b * N_ * C_ + h * HD_ + d;
  float s = 0.0f;
  for (int n = 0; n < N_; ++n) s += at[n] * vp[(size_t)n * C_];
  o[(size_t)(b * Q_ + j) * C_ + h * HD_ + d] = s;
}

// ---------------------------------------------------------------------------
// 11. agg = o @ out_proj_w.T + out_proj_b
// ---------------------------------------------------------------------------
__global__ void k_outproj(const float* __restrict__ o,
                          const float* __restrict__ opw,
                          const float* __restrict__ opb,
                          float* __restrict__ agg) {
  int idx = blockIdx.x * blockDim.x + threadIdx.x;
  if (idx >= B_ * Q_ * C_) return;
  int eo = idx & (C_ - 1);
  int t = idx >> 10;
  int j = t % Q_;
  int b = t / Q_;
  const float* op = o + (size_t)(b * Q_ + j) * C_;
  const float* w = opw + (size_t)eo * C_;
  float s = opb[eo];
  for (int c = 0; c < C_; ++c) s += op[c] * w[c];
  agg[idx] = s;
}

// ---------------------------------------------------------------------------
// host-side orchestration
// ---------------------------------------------------------------------------
extern "C" void kernel_launch(void* const* d_in, const int* in_sizes, int n_in,
                              void* d_out, int out_size, void* d_ws, size_t ws_size,
                              hipStream_t stream) {
  const int*   nodes   = (const int*)d_in[0];
  const int*   edges   = (const int*)d_in[1];
  const int*   types   = (const int*)d_in[2];
  const float* emb     = (const float*)d_in[3];
  const float* W       = (const float*)d_in[4];   // (L,R,C,C)
  const float* Wroot   = (const float*)d_in[5];   // (L,C,C)
  const float* bias    = (const float*)d_in[6];   // (L,C)
  const float* gprompt = (const float*)d_in[7];   // (Q,C)
  const float* ipw     = (const float*)d_in[8];   // (3C,C)
  const float* ipb     = (const float*)d_in[9];   // (3C)
  const float* opw     = (const float*)d_in[10];  // (C,C)
  const float* opb     = (const float*)d_in[11];  // (C)

  float* out_node = (float*)d_out;                // (B,N,C)
  float* out_agg  = out_node + BNC;               // (B,Q,C)

  char* wp = (char*)d_ws;
  float*  bufA = (float*)wp;  wp += BNC * 4;      // layer input / kbuf
  float*  bufB = (float*)wp;  wp += BNC * 4;      // layer output / vbuf
  float*  bufX = (float*)wp;  wp += BNC * 4;      // xt / scores
  __bf16* Abf  = (__bf16*)wp; wp += BNC * 2;      // bf16 activations
  __bf16* Wt   = (__bf16*)wp; wp += (size_t)C_ * C_ * 2;  // bf16 (N,K) weight
  int*    cnt  = (int*)wp;    wp += (size_t)B_ * N_ * R_ * 4;
  float*  inv  = (float*)wp;  wp += (size_t)B_ * N_ * R_ * 4;
  float*  qb   = (float*)wp;  wp += (size_t)Q_ * C_ * 4;
  float*  ob   = (float*)wp;

  // 1. gather embeddings
  k_gather<<<B_ * N_, 256, 0, stream>>>(nodes, emb, bufA);

  // 2. per-(dst, relation) counts -> inverse factors (shared by both layers)
  {
    int n = B_ * N_ * R_;
    k_zero_i<<<(n + 255) / 256, 256, 0, stream>>>(cnt, n);
    k_count<<<(B_ * E_ + 255) / 256, 256, 0, stream>>>(edges, types, cnt);
    k_inv<<<(n + 255) / 256, 256, 0, stream>>>(cnt, inv, n);
  }

  const long nAct4 = (long)(BNC / 4);
  const int  gAct  = (int)((nAct4 + 255) / 256);
  const int  gWT   = (C_ * C_) / 256;             // 4096 blocks
  dim3 ggrid(512 / 128, B_, C_ / 128);            // (4, 8, 8)

  // 3. RGCN layers
  const float* xin = bufA;
  for (int l = 0; l < L_; ++l) {
    float* xout = (l == 0) ? bufB : out_node;
    k_cvt4<<<gAct, 256, 0, stream>>>(xin, Abf, nAct4);
    // root: xout = xin @ Wroot[l] + bias[l]
    k_cvtT<<<gWT, 256, 0, stream>>>(Wroot + (size_t)l * C_ * C_, Wt);
    k_gemm_bf16<<<ggrid, 256, 0, stream>>>(Abf, Wt, bias + (size_t)l * C_, xout);
    // relations: xt = xin @ W[l,r]; scatter normalized messages into xout
    for (int r = 0; r < R_; ++r) {
      k_cvtT<<<gWT, 256, 0, stream>>>(W + ((size_t)l * R_ + r) * C_ * C_, Wt);
      k_gemm_bf16<<<ggrid, 256, 0, stream>>>(Abf, Wt, nullptr, bufX);
      k_scatter<<<dim3(E_, B_), 256, 0, stream>>>(bufX, edges, types, inv,
                                                  xout, r);
    }
    if (l == 0)
      k_relu<<<(int)((BNC + 255) / 256), 256, 0, stream>>>(xout, (long)BNC);
    xin = xout;
  }

  // 4. attention pooling
  k_cvt4<<<gAct, 256, 0, stream>>>(out_node, Abf, nAct4);
  // in_proj_w is stored (d, c) == (N, K): plain convert, no transpose
  k_cvt4<<<gWT, 256, 0, stream>>>(ipw + (size_t)C_ * C_, Wt,
                                  (long)(C_ * C_ / 4));
  k_gemm_bf16<<<ggrid, 256, 0, stream>>>(Abf, Wt, ipb + C_, bufA);      // k
  k_cvt4<<<gWT, 256, 0, stream>>>(ipw + 2 * (size_t)C_ * C_, Wt,
                                  (long)(C_ * C_ / 4));
  k_gemm_bf16<<<ggrid, 256, 0, stream>>>(Abf, Wt, ipb + 2 * C_, bufB);  // v

  k_qproj<<<(Q_ * C_ + 255) / 256, 256, 0, stream>>>(gprompt, ipw, ipb, qb);
  k_scores<<<(B_ * H_ * Q_ * N_ + 255) / 256, 256, 0, stream>>>(qb, bufA, bufX);
  k_softmax<<<(B_ * H_ * Q_ * 32 + 255) / 256, 256, 0, stream>>>(bufX);
  k_attnv<<<(B_ * H_ * Q_ * HD_ + 255) / 256, 256, 0, stream>>>(bufX, bufB, ob);
  k_outproj<<<(B_ * Q_ * C_ + 255) / 256, 256, 0, stream>>>(ob, opw, opb,
                                                            out_agg);
}
